// SwitchMoE_47699906789406
// MI455X (gfx1250) — matrix operand order repaired and checked by reading, not compiled, not run on hardware
//
#include <hip/hip_runtime.h>
#include <hip/hip_bf16.h>
#include <math.h>

// ---------------------------------------------------------------------------
// Switch-MoE (top-1) for gfx1250 / MI455X.
//   router -> histogram -> prefix -> scatter -> grouped GEMM1 (gelu) -> grouped GEMM2
// bf16 WMMA (v_wmma_f32_16x16x32_bf16), fp32 accumulate. wave32 everywhere.
// GEMM2 stages its bf16 A-tile with gfx1250 async-to-LDS copies (ASYNCcnt).
// ---------------------------------------------------------------------------

typedef __bf16 bf16;
typedef __attribute__((ext_vector_type(16))) bf16 bf16x16;
typedef __attribute__((ext_vector_type(8)))  bf16 bf16x8;
typedef __attribute__((ext_vector_type(4)))  bf16 bf16x4;
typedef __attribute__((ext_vector_type(8)))  float f32x8;
typedef __attribute__((ext_vector_type(4)))  float f32x4;

#define T_TOK 8192
#define D_EMB 1024
#define DFF   4096
#define NEXP  8
#define MT 128          // block tile M (tokens)
#define NT 128          // block tile N
#define KC 64           // K chunk staged in LDS

union Frag16 { bf16x16 v; bf16x8 h[2]; };

// ---------------------------------------------------------------------------
// tiny setup kernels
// ---------------------------------------------------------------------------
__global__ void moe_init(int* counts, int* cursors) {
    int i = threadIdx.x;
    if (i < NEXP) { counts[i] = 0; cursors[i] = 0; }
}

__global__ __launch_bounds__(256) void moe_router(
    const float* __restrict__ x, const float* __restrict__ gw,
    int* __restrict__ eidx, float* __restrict__ ew, int* __restrict__ counts)
{
    const int wave = threadIdx.x >> 5;
    const int lane = threadIdx.x & 31;
    const int t = blockIdx.x * 8 + wave;          // one wave per token
    const float* xr = x + (size_t)t * D_EMB;

    float acc[NEXP];
#pragma unroll
    for (int e = 0; e < NEXP; ++e) acc[e] = 0.0f;

    for (int k = lane; k < D_EMB; k += 32) {
        float xv = xr[k];
        const float* g = gw + (size_t)k * NEXP;   // 32B-aligned row of 8 floats
        f32x4 g0 = *(const f32x4*)g;
        f32x4 g1 = *(const f32x4*)(g + 4);
        acc[0] += xv * g0[0]; acc[1] += xv * g0[1];
        acc[2] += xv * g0[2]; acc[3] += xv * g0[3];
        acc[4] += xv * g1[0]; acc[5] += xv * g1[1];
        acc[6] += xv * g1[2]; acc[7] += xv * g1[3];
    }
#pragma unroll
    for (int off = 16; off > 0; off >>= 1)
#pragma unroll
        for (int e = 0; e < NEXP; ++e)
            acc[e] += __shfl_down(acc[e], off);

    if (lane == 0) {
        float mx = acc[0]; int am = 0;
#pragma unroll
        for (int e = 1; e < NEXP; ++e)
            if (acc[e] > mx) { mx = acc[e]; am = e; }   // first-max like jnp.argmax
        float s = 0.0f;
#pragma unroll
        for (int e = 0; e < NEXP; ++e) s += expf(acc[e] - mx);
        eidx[t] = am;
        ew[t] = 1.0f / s;                                // exp(mx-mx)/sum
        atomicAdd(&counts[am], 1);
    }
}

__global__ void moe_prefix(const int* __restrict__ counts, int* __restrict__ offs) {
    if (threadIdx.x == 0) {
        int run = 0;
        for (int e = 0; e < NEXP; ++e) { offs[e] = run; run += counts[e]; }
    }
}

__global__ __launch_bounds__(256) void moe_scatter(
    const int* __restrict__ eidx, const int* __restrict__ offs,
    int* __restrict__ cursors, int* __restrict__ tlist)
{
    int t = blockIdx.x * 256 + threadIdx.x;
    if (t >= T_TOK) return;
    int e = eidx[t];
    int pos = offs[e] + atomicAdd(&cursors[e], 1);
    tlist[pos] = t;
}

// ---------------------------------------------------------------------------
// GEMM1: h[p, :DFF] = gelu( x[tok(p), :] @ w1[e] + b1[e] ), bf16 out
// grid: (DFF/NT, T/MT, NEXP); blocks past the expert's row count exit.
// ---------------------------------------------------------------------------
__global__ __launch_bounds__(256) void moe_gemm1(
    const float* __restrict__ x, const float* __restrict__ w1, const float* __restrict__ b1,
    const int* __restrict__ counts, const int* __restrict__ offs,
    const int* __restrict__ tlist, bf16* __restrict__ hbuf)
{
    const int e = blockIdx.z;
    const int cnt = counts[e];
    const int mBase = blockIdx.y * MT;
    if (mBase >= cnt) return;
    const int segStart = offs[e];
    const int n0 = blockIdx.x * NT;
    const int rowsValid = min(MT, cnt - mBase);

    __shared__ bf16 Alds[MT][KC];     // tokens  [m][k]
    __shared__ bf16 Btl[NT][KC];      // weights transposed [n][k]
    __shared__ int  tlds[MT];

    const int tid = threadIdx.x;
    if (tid < MT) tlds[tid] = (tid < rowsValid) ? tlist[segStart + mBase + tid] : -1;
    __syncthreads();

    const int lane = tid & 31;
    const int wave = tid >> 5;
    const int mW = (wave >> 2) * 64;  // 2 waves along M
    const int nW = (wave & 3) * 32;   // 4 waves along N
    const int hh = lane >> 4;
    const int ln = lane & 15;

    f32x8 acc[4][2];
#pragma unroll
    for (int i = 0; i < 4; ++i)
#pragma unroll
        for (int j = 0; j < 2; ++j)
            acc[i][j] = (f32x8){0.f,0.f,0.f,0.f,0.f,0.f,0.f,0.f};

    const float* w1e = w1 + (size_t)e * D_EMB * DFF;

    for (int k0 = 0; k0 < D_EMB; k0 += KC) {
        // stage A: gathered token rows, fp32 -> bf16
#pragma unroll
        for (int s = 0; s < 8; ++s) {
            int elt = tid * 4 + s * 1024;
            int m = elt >> 6, k = elt & 63;
            int t = tlds[m];
            f32x4 v = (f32x4){0.f, 0.f, 0.f, 0.f};
            if (t >= 0) v = *(const f32x4*)(x + (size_t)t * D_EMB + k0 + k);
            bf16x4 bv;
            bv[0] = (bf16)v[0]; bv[1] = (bf16)v[1];
            bv[2] = (bf16)v[2]; bv[3] = (bf16)v[3];
            *(bf16x4*)(&Alds[m][k]) = bv;
        }
        // stage B: fp32 -> bf16, transposed to [n][k] so frag loads are b128
#pragma unroll
        for (int s = 0; s < 8; ++s) {
            int elt = tid * 4 + s * 1024;
            int kb = elt >> 7, nn = elt & 127;
            const float* src = w1e + (size_t)(k0 + kb) * DFF + n0 + nn;
            if (s == 0 && k0 + KC < D_EMB)
                __builtin_prefetch(src + (size_t)KC * DFF, 0, 1);  // global_prefetch_b8
            f32x4 v = *(const f32x4*)src;
            Btl[nn + 0][kb] = (bf16)v[0];
            Btl[nn + 1][kb] = (bf16)v[1];
            Btl[nn + 2][kb] = (bf16)v[2];
            Btl[nn + 3][kb] = (bf16)v[3];
        }
        __syncthreads();
#pragma unroll
        for (int kk = 0; kk < KC; kk += 32) {
            Frag16 bfr[2];
#pragma unroll
            for (int j = 0; j < 2; ++j) {
                bfr[j].h[0] = *(const bf16x8*)(&Btl[nW + j*16 + ln][kk + hh*8]);
                bfr[j].h[1] = *(const bf16x8*)(&Btl[nW + j*16 + ln][kk + 16 + hh*8]);
            }
#pragma unroll
            for (int i = 0; i < 4; ++i) {
                Frag16 af;
                af.h[0] = *(const bf16x8*)(&Alds[mW + i*16 + ln][kk + hh*8]);
                af.h[1] = *(const bf16x8*)(&Alds[mW + i*16 + ln][kk + 16 + hh*8]);
#pragma unroll
                for (int j = 0; j < 2; ++j)
                    acc[i][j] = __builtin_amdgcn_wmma_f32_16x16x32_bf16(
                        false, af.v, false, bfr[j].v, (short)0, acc[i][j], false, false);
            }
        }
        __syncthreads();
    }

    // epilogue: bias + exact GeLU -> bf16 h (grouped-row order)
    float bias[2];
#pragma unroll
    for (int j = 0; j < 2; ++j)
        bias[j] = b1[(size_t)e * DFF + n0 + nW + j*16 + ln];
#pragma unroll
    for (int i = 0; i < 4; ++i)
#pragma unroll
        for (int j = 0; j < 2; ++j)
#pragma unroll
            for (int r = 0; r < 8; ++r) {
                int ml = mW + i*16 + hh*8 + r;
                if (ml < rowsValid) {
                    float v = acc[i][j][r] + bias[j];
                    float g = 0.5f * v * (1.0f + erff(v * 0.70710678118654752f));
                    size_t p = (size_t)(segStart + mBase + ml);
                    hbuf[p * DFF + n0 + nW + j*16 + ln] = (bf16)g;
                }
            }
}

// ---------------------------------------------------------------------------
// GEMM2: out[tok(p), :] = ew[tok(p)] * ( h[p, :] @ w2[e] + b2[e] )
// grid: (D_EMB/NT, T/MT, NEXP)
// A tile (bf16 h rows) staged with async-to-LDS copies: no VALU/VGPR traffic,
// overlapped with the fp32->bf16 conversion of the B tile.
// ---------------------------------------------------------------------------
__global__ __launch_bounds__(256) void moe_gemm2(
    const bf16* __restrict__ hbuf, const float* __restrict__ w2, const float* __restrict__ b2,
    const float* __restrict__ ew, const int* __restrict__ counts, const int* __restrict__ offs,
    const int* __restrict__ tlist, float* __restrict__ out)
{
    const int e = blockIdx.z;
    const int cnt = counts[e];
    const int mBase = blockIdx.y * MT;
    if (mBase >= cnt) return;
    const int segStart = offs[e];
    const int n0 = blockIdx.x * NT;
    const int rowsValid = min(MT, cnt - mBase);

    __shared__ bf16 Alds[MT][KC];
    __shared__ bf16 Btl[NT][KC];
    __shared__ int  tlds[MT];

    const int tid = threadIdx.x;
    if (tid < MT) tlds[tid] = (tid < rowsValid) ? tlist[segStart + mBase + tid] : -1;
    __syncthreads();

    const int lane = tid & 31;
    const int wave = tid >> 5;
    const int mW = (wave >> 2) * 64;
    const int nW = (wave & 3) * 32;
    const int hh = lane >> 4;
    const int ln = lane & 15;

    f32x8 acc[4][2];
#pragma unroll
    for (int i = 0; i < 4; ++i)
#pragma unroll
        for (int j = 0; j < 2; ++j)
            acc[i][j] = (f32x8){0.f,0.f,0.f,0.f,0.f,0.f,0.f,0.f};

    const float* w2e = w2 + (size_t)e * DFF * D_EMB;

    // per-thread A-tile async copy slots: 4 x 16B per thread = 16KB tile
    // chunk c = tid + s*256 ; row m = c/8 ; col k = (c%8)*8 bf16 (16B aligned)
    unsigned ldsA[4];
    int      rowA[4];
    int      colA[4];
#pragma unroll
    for (int s = 0; s < 4; ++s) {
        int c = tid + s * 256;
        int m = c >> 3;
        int k = (c & 7) * 8;
        rowA[s] = segStart + mBase + ((m < rowsValid) ? m : 0);
        colA[s] = k;
        ldsA[s] = (unsigned)(uintptr_t)&Alds[m][k];   // low 32 bits = LDS offset
    }

    for (int k0 = 0; k0 < DFF; k0 += KC) {
        // stage A: bf16 h rows, async global->LDS (ASYNCcnt), no VGPR data path
#pragma unroll
        for (int s = 0; s < 4; ++s) {
            unsigned long long ga =
                (unsigned long long)(uintptr_t)(hbuf + (size_t)rowA[s] * DFF + k0 + colA[s]);
            asm volatile("global_load_async_to_lds_b128 %0, %1, off"
                         :: "v"(ldsA[s]), "v"(ga) : "memory");
        }
        // stage B: fp32 -> bf16, transposed (overlaps with async A copy)
#pragma unroll
        for (int s = 0; s < 8; ++s) {
            int elt = tid * 4 + s * 1024;
            int kb = elt >> 7, nn = elt & 127;
            const float* src = w2e + (size_t)(k0 + kb) * D_EMB + n0 + nn;
            if (s == 0 && k0 + KC < DFF)
                __builtin_prefetch(src + (size_t)KC * D_EMB, 0, 1);
            f32x4 v = *(const f32x4*)src;
            Btl[nn + 0][kb] = (bf16)v[0];
            Btl[nn + 1][kb] = (bf16)v[1];
            Btl[nn + 2][kb] = (bf16)v[2];
            Btl[nn + 3][kb] = (bf16)v[3];
        }
        asm volatile("s_wait_asynccnt 0x0" ::: "memory");
        __syncthreads();
#pragma unroll
        for (int kk = 0; kk < KC; kk += 32) {
            Frag16 bfr[2];
#pragma unroll
            for (int j = 0; j < 2; ++j) {
                bfr[j].h[0] = *(const bf16x8*)(&Btl[nW + j*16 + ln][kk + hh*8]);
                bfr[j].h[1] = *(const bf16x8*)(&Btl[nW + j*16 + ln][kk + 16 + hh*8]);
            }
#pragma unroll
            for (int i = 0; i < 4; ++i) {
                Frag16 af;
                af.h[0] = *(const bf16x8*)(&Alds[mW + i*16 + ln][kk + hh*8]);
                af.h[1] = *(const bf16x8*)(&Alds[mW + i*16 + ln][kk + 16 + hh*8]);
#pragma unroll
                for (int j = 0; j < 2; ++j)
                    acc[i][j] = __builtin_amdgcn_wmma_f32_16x16x32_bf16(
                        false, af.v, false, bfr[j].v, (short)0, acc[i][j], false, false);
            }
        }
        __syncthreads();
    }

    // epilogue: bias, scale by routing weight, scatter to token row
    float bias[2];
#pragma unroll
    for (int j = 0; j < 2; ++j)
        bias[j] = b2[(size_t)e * D_EMB + n0 + nW + j*16 + ln];
#pragma unroll
    for (int i = 0; i < 4; ++i)
#pragma unroll
        for (int j = 0; j < 2; ++j)
#pragma unroll
            for (int r = 0; r < 8; ++r) {
                int ml = mW + i*16 + hh*8 + r;
                if (ml < rowsValid) {
                    int t = tlds[ml];
                    float wgt = ew[t];
                    out[(size_t)t * D_EMB + n0 + nW + j*16 + ln] =
                        wgt * (acc[i][j][r] + bias[j]);
                }
            }
}

// ---------------------------------------------------------------------------
// host launcher
// ---------------------------------------------------------------------------
extern "C" void kernel_launch(void* const* d_in, const int* in_sizes, int n_in,
                              void* d_out, int out_size, void* d_ws, size_t ws_size,
                              hipStream_t stream) {
    const float* x  = (const float*)d_in[0];
    const float* gw = (const float*)d_in[1];
    const float* w1 = (const float*)d_in[2];
    const float* b1 = (const float*)d_in[3];
    const float* w2 = (const float*)d_in[4];
    const float* b2 = (const float*)d_in[5];
    float* out = (float*)d_out;

    char* ws = (char*)d_ws;
    size_t off = 0;
    auto take = [&](size_t bytes) -> char* {
        char* p = ws + off;
        off = (off + bytes + 255) & ~(size_t)255;
        return p;
    };
    int*   eidx    = (int*)  take((size_t)T_TOK * 4);
    float* ew      = (float*)take((size_t)T_TOK * 4);
    int*   counts  = (int*)  take(NEXP * 4);
    int*   offs    = (int*)  take(NEXP * 4);
    int*   cursors = (int*)  take(NEXP * 4);
    int*   tlist   = (int*)  take((size_t)T_TOK * 4);
    bf16*  hbuf    = (bf16*) take((size_t)T_TOK * DFF * 2);
    (void)ws_size; (void)in_sizes; (void)n_in; (void)out_size;

    moe_init<<<1, 32, 0, stream>>>(counts, cursors);
    moe_router<<<T_TOK / 8, 256, 0, stream>>>(x, gw, eidx, ew, counts);
    moe_prefix<<<1, 1, 0, stream>>>(counts, offs);
    moe_scatter<<<T_TOK / 256, 256, 0, stream>>>(eidx, offs, cursors, tlist);
    moe_gemm1<<<dim3(DFF / NT, T_TOK / MT, NEXP), 256, 0, stream>>>(
        x, w1, b1, counts, offs, tlist, hbuf);
    moe_gemm2<<<dim3(D_EMB / NT, T_TOK / MT, NEXP), 256, 0, stream>>>(
        hbuf, w2, b2, ew, counts, offs, tlist, out);
}
